// Model_52381421142395
// MI455X (gfx1250) — matrix operand order repaired and checked by reading, not compiled, not run on hardware
//
#include <hip/hip_runtime.h>
#include <hip/hip_bf16.h>
#include <math.h>
#include <stdint.h>

// ---------------- problem constants ----------------
#define B_   64
#define N_   1024
#define F_   195
#define FP_  224      // F padded to multiple of 32 (7 K-steps)
#define H_   128
#define O_   64
#define K1_  512
#define K2_  256

typedef __bf16 bf16;
typedef __bf16  v16bf  __attribute__((ext_vector_type(16)));
typedef __bf16  bf16x4 __attribute__((ext_vector_type(4)));
typedef float   v8f    __attribute__((ext_vector_type(8)));
typedef int     v2i    __attribute__((ext_vector_type(2)));

// ---------------------------------------------------------------------
// Async global->LDS copy (CDNA5 ASYNCcnt path).
// Probe-confirmed signature: (v2i32 __device__*, v2i32 __shared__*, i32, i32)
// Cast through uintptr_t to produce the address-space-qualified pointers.
// ---------------------------------------------------------------------
#if defined(__gfx1250__) && __has_builtin(__builtin_amdgcn_global_load_async_to_lds_b64)
#define ASYNC_LDS 1
#else
#define ASYNC_LDS 0
#endif

#if ASYNC_LDS
typedef __attribute__((address_space(1))) v2i* as1_v2i_ptr;
typedef __attribute__((address_space(3))) v2i* as3_v2i_ptr;
#endif

__device__ __forceinline__ void lds_copy_b64(void* lds, const void* g) {
#if ASYNC_LDS
  __builtin_amdgcn_global_load_async_to_lds_b64(
      (as1_v2i_ptr)(uintptr_t)g, (as3_v2i_ptr)(uintptr_t)lds, 0, 0);
#else
  *(bf16x4*)lds = *(const bf16x4*)g;
#endif
}

__device__ __forceinline__ void lds_copy_fence() {
#if ASYNC_LDS && __has_builtin(__builtin_amdgcn_s_wait_asynccnt)
  __builtin_amdgcn_s_wait_asynccnt(0);
#endif
}

// =====================================================================
// Kernel 1: center rows, cast to bf16 (padded), compute 1/||xc|| per row
// one wave (32 lanes) per (b,n) row; 8 rows per 256-thread block
// =====================================================================
__global__ void k_prep(const float* __restrict__ in, bf16* __restrict__ xbuf,
                       float* __restrict__ invn) {
  const int lane = threadIdx.x & 31;
  const int row  = blockIdx.x * 8 + (threadIdx.x >> 5);
  if (row >= B_ * N_) return;
  const float* src = in + (size_t)row * F_;

  float s = 0.f;
  for (int k = lane; k < F_; k += 32) s += src[k];
  for (int o = 16; o; o >>= 1) s += __shfl_down(s, o, 32);
  s = __shfl(s, 0, 32);
  const float mu = s / (float)F_;

  float q = 0.f;
  for (int k = lane; k < F_; k += 32) { float v = src[k] - mu; q += v * v; }
  for (int o = 16; o; o >>= 1) q += __shfl_down(q, o, 32);
  q = __shfl(q, 0, 32);
  if (lane == 0) invn[row] = rsqrtf(fmaxf(q, 1e-12f));

  bf16* dst = xbuf + (size_t)row * FP_;
  for (int k = lane; k < FP_; k += 32)
    dst[k] = (bf16)(k < F_ ? src[k] - mu : 0.f);
}

// =====================================================================
// Kernel 2: cast raw input to padded bf16 (reuses xbuf after corr done)
// =====================================================================
__global__ void k_cast_raw(const float* __restrict__ in, bf16* __restrict__ xbuf) {
  size_t t = (size_t)blockIdx.x * blockDim.x + threadIdx.x;
  if (t >= (size_t)B_ * N_ * FP_) return;
  int k = (int)(t % FP_);
  size_t row = t / FP_;
  xbuf[t] = (bf16)(k < F_ ? in[row * F_ + k] : 0.f);
}

// ---------------------------------------------------------------------
// Fragment-order LDS staging (WMMA 16x16x32 bf16 layouts):
//   A (16x32):  lane = (m&15) + 16*((k>>3)&1),  e = (k&7) + 8*((k>>4)&1)
//   B (32x16):  lane = (n&15) + 16*((k>>3)&1),  e = (k&7) + 8*((k>>4)&1)
// 4 consecutive k stay contiguous in e -> b64 copies, b128 reads.
// ---------------------------------------------------------------------

// =====================================================================
// Kernel 3: correlation adjacency via WMMA bf16 (symmetric: only C >= R
// tiles computed; epilogue sets both (m,n) and (n,m) bits)
// =====================================================================
__global__ void k_corr_wmma(const bf16* __restrict__ Xc, const float* __restrict__ invn,
                            unsigned* __restrict__ Abits) {
  __shared__ __align__(32) bf16 fragA[4][32][16];   // [row-subtile][lane][e]
  __shared__ __align__(32) bf16 fragB[4][32][16];   // [col-subtile][lane][e]
  if (blockIdx.y < blockIdx.x) return;              // symmetry: skip lower tiles
  const int b    = blockIdx.z;
  const int R    = blockIdx.x * 64;
  const int C    = blockIdx.y * 64;
  const int lane = threadIdx.x & 31;
  const int wave = threadIdx.x >> 5;
  const bf16* Xb = Xc + (size_t)b * N_ * FP_;

  v8f acc[4];
  #pragma unroll
  for (int j = 0; j < 4; ++j) acc[j] = (v8f)0.f;

  for (int kk = 0; kk < FP_; kk += 32) {
    __syncthreads();
    // A tile: rows R..R+63, k kk..kk+31 (async b64 per 4 k's)
    for (int j = threadIdx.x; j < 64 * 8; j += blockDim.x) {
      int r  = j >> 3;
      int k0 = (j & 7) << 2;
      int ln = (r & 15) + (((k0 >> 3) & 1) << 4);
      int e0 = (k0 & 7) + ((k0 >> 4) << 3);
      lds_copy_b64(&fragA[r >> 4][ln][e0], &Xb[(size_t)(R + r) * FP_ + kk + k0]);
    }
    // B tile: cols C..C+63 are also Xc rows -> contiguous in k
    for (int j = threadIdx.x; j < 64 * 8; j += blockDim.x) {
      int n  = j >> 3;
      int k0 = (j & 7) << 2;
      int ln = (n & 15) + (((k0 >> 3) & 1) << 4);
      lds_copy_b64(&fragB[n >> 4][ln][(k0 & 7) + ((k0 >> 4) << 3)], &Xb[(size_t)(C + n) * FP_ + kk + k0]);
    }
    lds_copy_fence();
    __syncthreads();

    v16bf af = *(const v16bf*)fragA[wave][lane];
    #pragma unroll
    for (int j = 0; j < 4; ++j) {
      v16bf bfg = *(const v16bf*)fragB[j][lane];
      acc[j] = __builtin_amdgcn_wmma_f32_16x16x32_bf16(
          false, af, false, bfg, (short)0, acc[j], false, false);
    }
  }

  // epilogue: normalize, threshold, set (m,n) and mirrored (n,m) bits
  const int nn = lane & 15;
  #pragma unroll
  for (int j = 0; j < 4; ++j) {
    #pragma unroll
    for (int r = 0; r < 8; ++r) {
      int mrow = R + 16 * wave + r + ((lane >> 4) << 3);
      int ncol = C + 16 * j + nn;
      float c = acc[j][r] * invn[b * N_ + mrow] * invn[b * N_ + ncol];
      if (c > 0.5f) {
        atomicOr(&Abits[((size_t)b * N_ + mrow) * (N_ / 32) + (ncol >> 5)],
                 1u << (ncol & 31));
        atomicOr(&Abits[((size_t)b * N_ + ncol) * (N_ / 32) + (mrow >> 5)],
                 1u << (mrow & 31));
      }
    }
  }
}

// =====================================================================
// Kernel 4: degree = popcount of adjacency bitmask row
// =====================================================================
__global__ void k_popc(const unsigned* __restrict__ bits, float* __restrict__ deg,
                       int rows, int wpr) {
  int t = blockIdx.x * blockDim.x + threadIdx.x;
  if (t >= B_ * rows) return;
  const unsigned* r = bits + (size_t)t * wpr;
  int s = 0;
  for (int w = 0; w < wpr; ++w) s += __popc(r[w]);
  deg[t] = (float)s;
}

// =====================================================================
// Kernel 5: adjacency(bits) x features(bf16) via WMMA -> f32 out
// bits expanded to bf16 {0,1} fragment tiles in LDS; optional mean
// =====================================================================
__global__ void k_agg_wmma(const unsigned* __restrict__ Abits, int wpr, int rows,
                           const bf16* __restrict__ X, int ldx, int xcols,
                           float* __restrict__ out, int ldo,
                           const float* __restrict__ deg) {
  __shared__ __align__(32) bf16 fragA[4][32][16];
  __shared__ __align__(32) bf16 fragB[4][32][16];
  const int b    = blockIdx.z;
  const int R    = blockIdx.x * 64;
  const int C    = blockIdx.y * 64;
  const int lane = threadIdx.x & 31;
  const int wave = threadIdx.x >> 5;

  v8f acc[4];
  #pragma unroll
  for (int j = 0; j < 4; ++j) acc[j] = (v8f)0.f;

  for (int kk = 0; kk < rows; kk += 32) {
    __syncthreads();
    // A tile: expand 4 bits -> bf16x4, one b64 store
    for (int j = threadIdx.x; j < 64 * 8; j += blockDim.x) {
      int r  = j >> 3;
      int k0 = (j & 7) << 2;
      unsigned w = Abits[((size_t)b * rows + R + r) * wpr + ((kk + k0) >> 5)];
      bf16x4 v;
      #pragma unroll
      for (int t = 0; t < 4; ++t)
        v[t] = (bf16)(float)((w >> ((kk + k0 + t) & 31)) & 1u);
      int ln = (r & 15) + (((k0 >> 3) & 1) << 4);
      int e0 = (k0 & 7) + ((k0 >> 4) << 3);
      *(bf16x4*)&fragA[r >> 4][ln][e0] = v;
    }
    // B tile: 4 k's per thread (coalesced across lanes), one b64 store
    for (int j = threadIdx.x; j < 512; j += blockDim.x) {
      int n  = j & 63;
      int k0 = (j >> 6) << 2;
      bf16x4 v;
      if (C + n < xcols) {
        #pragma unroll
        for (int t = 0; t < 4; ++t)
          v[t] = X[((size_t)b * rows + kk + k0 + t) * ldx + C + n];
      } else {
        #pragma unroll
        for (int t = 0; t < 4; ++t) v[t] = (bf16)0.f;
      }
      int ln = (n & 15) + (((k0 >> 3) & 1) << 4);
      *(bf16x4*)&fragB[n >> 4][ln][(k0 & 7) + ((k0 >> 4) << 3)] = v;
    }
    if (kk + 32 < rows)
      __builtin_prefetch(&X[((size_t)b * rows + kk + 32 + (threadIdx.x >> 2)) * ldx + C], 0, 1);
    __syncthreads();

    v16bf af = *(const v16bf*)fragA[wave][lane];
    #pragma unroll
    for (int j = 0; j < 4; ++j) {
      v16bf bfg = *(const v16bf*)fragB[j][lane];
      acc[j] = __builtin_amdgcn_wmma_f32_16x16x32_bf16(
          false, af, false, bfg, (short)0, acc[j], false, false);
    }
  }

  const int nn = lane & 15;
  #pragma unroll
  for (int j = 0; j < 4; ++j) {
    #pragma unroll
    for (int r = 0; r < 8; ++r) {
      int mrow = R + 16 * wave + r + ((lane >> 4) << 3);
      int ncol = C + 16 * j + nn;
      if (ncol < ldo) {
        float v = acc[j][r];
        if (deg) v /= fmaxf(deg[b * rows + mrow], 1.f);
        out[((size_t)b * rows + mrow) * ldo + ncol] = v;
      }
    }
  }
}

// =====================================================================
// Kernel 6: SAGE linear: relu(Wl*agg + bl + Wr*x) -> bf16
// =====================================================================
__global__ void k_sage_lin(const float* __restrict__ agg, int ldo,
                           const float* __restrict__ xf, const bf16* __restrict__ xb, int ldx,
                           const float* __restrict__ Wl, const float* __restrict__ bl,
                           const float* __restrict__ Wr,
                           int F, int H, int rows, bf16* __restrict__ out) {
  int t = blockIdx.x * blockDim.x + threadIdx.x;
  if (t >= B_ * rows * H) return;
  int h = t % H;
  size_t rn = (size_t)(t / H);                 // b*rows + n
  const float* ag = agg + rn * ldo;
  float acc = bl[h];
  if (xf) {
    const float* x = xf + rn * ldx;
    for (int f = 0; f < F; ++f)
      acc += ag[f] * Wl[h * F + f] + x[f] * Wr[h * F + f];
  } else {
    const bf16* x = xb + rn * ldx;
    for (int f = 0; f < F; ++f)
      acc += ag[f] * Wl[h * F + f] + (float)x[f] * Wr[h * F + f];
  }
  out[rn * H + h] = (bf16)fmaxf(acc, 0.f);
}

// =====================================================================
// Kernel 7: pooling score: Wrel*agg + brel + Wroot*x  -> f32 [B*rows]
// =====================================================================
__global__ void k_score(const float* __restrict__ agg, int ldo,
                        const bf16* __restrict__ x, int ldx,
                        const float* __restrict__ Wrel, const float* __restrict__ brel,
                        const float* __restrict__ Wroot,
                        int F, int rows, float* __restrict__ score) {
  int t = blockIdx.x * blockDim.x + threadIdx.x;
  if (t >= B_ * rows) return;
  const float* ag = agg + (size_t)t * ldo;
  const bf16*  xr = x + (size_t)t * ldx;
  float s = brel[0];
  for (int f = 0; f < F; ++f)
    s += ag[f] * Wrel[f] + (float)xr[f] * Wroot[f];
  score[t] = s;
}

// =====================================================================
// Kernel 8: per-graph top-k (bitonic sort, descending), gather x*tanh(s),
// gather adjacency bitmask (if wprOut > 0). One 512-thread block / graph.
// =====================================================================
__global__ void k_topk(const float* __restrict__ score, int n, int k,
                       const bf16* __restrict__ xin, int F, bf16* __restrict__ xg,
                       const unsigned* __restrict__ AbitsIn, int wprIn,
                       unsigned* __restrict__ AbitsOut, int wprOut) {
  __shared__ float keys[1024];
  __shared__ int   vals[1024];
  const int b = blockIdx.x;
  const int tid = threadIdx.x;

  for (int i = tid; i < n; i += blockDim.x) {
    keys[i] = score[b * n + i];
    vals[i] = i;
  }
  for (unsigned size = 2; size <= (unsigned)n; size <<= 1) {
    for (unsigned stride = size >> 1; stride > 0; stride >>= 1) {
      __syncthreads();
      for (unsigned i = tid; i < (unsigned)n; i += blockDim.x) {
        unsigned j = i ^ stride;
        if (j > i) {
          bool desc = ((i & size) == 0);
          float ki = keys[i], kj = keys[j];
          if (desc ? (ki < kj) : (ki > kj)) {
            keys[i] = kj; keys[j] = ki;
            int tv = vals[i]; vals[i] = vals[j]; vals[j] = tv;
          }
        }
      }
    }
  }
  __syncthreads();

  if (tid < k) {
    int row = vals[tid];
    float sc = tanhf(keys[tid]);
    const bf16* src = xin + ((size_t)b * n + row) * F;
    bf16* dst = xg + ((size_t)b * k + tid) * F;
    for (int f = 0; f < F; ++f) dst[f] = (bf16)((float)src[f] * sc);
  }
  __syncthreads();

  if (wprOut > 0) {
    for (int w = tid; w < k * wprOut; w += blockDim.x) {
      int i = w / wprOut, wj = w % wprOut;
      const unsigned* src = AbitsIn + ((size_t)b * n + vals[i]) * wprIn;
      unsigned bits = 0;
      for (int t2 = 0; t2 < 32; ++t2) {
        int col = vals[wj * 32 + t2];
        bits |= ((src[col >> 5] >> (col & 31)) & 1u) << t2;
      }
      AbitsOut[((size_t)b * k + i) * wprOut + wj] = bits;
    }
  }
}

// =====================================================================
// Kernel 9: global mean pool + MLP head (fc/bn/relu x2, fc3, softmax)
// =====================================================================
__global__ void k_final(const bf16* __restrict__ x2g,
                        const float* fc1_w, const float* fc1_b,
                        const float* g1, const float* b1, const float* m1, const float* v1,
                        const float* fc2_w, const float* fc2_b,
                        const float* g2, const float* b2, const float* m2, const float* v2,
                        const float* fc3_w, const float* fc3_b,
                        float* __restrict__ out) {
  __shared__ float pool[O_];
  __shared__ float h1[512];
  __shared__ float h2[256];
  const int b = blockIdx.x;
  const int tid = threadIdx.x;

  if (tid < O_) {
    float s = 0.f;
    for (int r = 0; r < K2_; ++r)
      s += (float)x2g[((size_t)b * K2_ + r) * O_ + tid];
    pool[tid] = s / (float)K2_;
  }
  __syncthreads();

  {
    float v = fc1_b[tid];
    for (int f = 0; f < O_; ++f) v += pool[f] * fc1_w[tid * O_ + f];
    v = fmaxf(v, 0.f);
    h1[tid] = g1[tid] * (v - m1[tid]) * rsqrtf(v1[tid] + 1e-5f) + b1[tid];
  }
  __syncthreads();

  if (tid < 256) {
    float v = fc2_b[tid];
    for (int f = 0; f < 512; ++f) v += h1[f] * fc2_w[tid * 512 + f];
    v = fmaxf(v, 0.f);
    h2[tid] = g2[tid] * (v - m2[tid]) * rsqrtf(v2[tid] + 1e-5f) + b2[tid];
  }
  __syncthreads();

  if (tid == 0) {
    float z0 = fc3_b[0], z1 = fc3_b[1];
    for (int f = 0; f < 256; ++f) {
      z0 += h2[f] * fc3_w[f];
      z1 += h2[f] * fc3_w[256 + f];
    }
    float mx = fmaxf(z0, z1);
    float e0 = expf(z0 - mx), e1 = expf(z1 - mx);
    out[b * 2 + 0] = e0 / (e0 + e1);
    out[b * 2 + 1] = e1 / (e0 + e1);
  }
}

// =====================================================================
// host: workspace layout + launch sequence
// =====================================================================
static constexpr size_t alup(size_t x) { return (x + 255) & ~(size_t)255; }

extern "C" void kernel_launch(void* const* d_in, const int* in_sizes, int n_in,
                              void* d_out, int out_size, void* d_ws, size_t ws_size,
                              hipStream_t stream) {
  (void)in_sizes; (void)n_in; (void)out_size; (void)ws_size;
  const float* input   = (const float*)d_in[0];
  const float* W1l     = (const float*)d_in[1];
  const float* b1l     = (const float*)d_in[2];
  const float* W1r     = (const float*)d_in[3];
  const float* Wp1rel  = (const float*)d_in[4];
  const float* bp1rel  = (const float*)d_in[5];
  const float* Wp1root = (const float*)d_in[6];
  const float* W2l     = (const float*)d_in[7];
  const float* b2l     = (const float*)d_in[8];
  const float* W2r     = (const float*)d_in[9];
  const float* Wp2rel  = (const float*)d_in[10];
  const float* bp2rel  = (const float*)d_in[11];
  const float* Wp2root = (const float*)d_in[12];
  const float* fc1_w   = (const float*)d_in[13];
  const float* fc1_b   = (const float*)d_in[14];
  const float* bn1_g   = (const float*)d_in[15];
  const float* bn1_b   = (const float*)d_in[16];
  const float* bn1_m   = (const float*)d_in[17];
  const float* bn1_v   = (const float*)d_in[18];
  const float* fc2_w   = (const float*)d_in[19];
  const float* fc2_b   = (const float*)d_in[20];
  const float* bn2_g   = (const float*)d_in[21];
  const float* bn2_b   = (const float*)d_in[22];
  const float* bn2_m   = (const float*)d_in[23];
  const float* bn2_v   = (const float*)d_in[24];
  const float* fc3_w   = (const float*)d_in[25];
  const float* fc3_b   = (const float*)d_in[26];

  char* ws = (char*)d_ws;
  size_t off = 0;
  auto carve = [&](size_t bytes) { char* p = ws + off; off += alup(bytes); return p; };

  bf16*     xbuf  = (bf16*)    carve((size_t)B_ * N_ * FP_ * 2);   // xc, later raw bf16
  float*    invn  = (float*)   carve((size_t)B_ * N_ * 4);
  unsigned* Ab1   = (unsigned*)carve((size_t)B_ * N_ * (N_ / 32) * 4);
  float*    deg1  = (float*)   carve((size_t)B_ * N_ * 4);
  float*    agg   = (float*)   carve((size_t)B_ * N_ * FP_ * 4);   // shared f32 agg buffer
  bf16*     x1    = (bf16*)    carve((size_t)B_ * N_ * H_ * 2);
  float*    score = (float*)   carve((size_t)B_ * N_ * 4);
  bf16*     x1g   = (bf16*)    carve((size_t)B_ * K1_ * H_ * 2);
  unsigned* Ab2   = (unsigned*)carve((size_t)B_ * K1_ * (K1_ / 32) * 4);
  float*    deg2  = (float*)   carve((size_t)B_ * K1_ * 4);
  bf16*     x2    = (bf16*)    carve((size_t)B_ * K1_ * O_ * 2);
  bf16*     x2g   = (bf16*)    carve((size_t)B_ * K2_ * O_ * 2);

  (void)hipMemsetAsync(Ab1, 0, (size_t)B_ * N_ * (N_ / 32) * 4, stream);

  k_prep<<<(B_ * N_ + 7) / 8, 256, 0, stream>>>(input, xbuf, invn);

  k_corr_wmma<<<dim3(N_ / 64, N_ / 64, B_), 128, 0, stream>>>(xbuf, invn, Ab1);

  k_popc<<<(B_ * N_ + 255) / 256, 256, 0, stream>>>(Ab1, deg1, N_, N_ / 32);

  k_cast_raw<<<(int)(((size_t)B_ * N_ * FP_ + 255) / 256), 256, 0, stream>>>(input, xbuf);

  k_agg_wmma<<<dim3(N_ / 64, 4, B_), 128, 0, stream>>>(Ab1, N_ / 32, N_,
      xbuf, FP_, FP_, agg, FP_, deg1);

  k_sage_lin<<<(B_ * N_ * H_ + 255) / 256, 256, 0, stream>>>(agg, FP_,
      input, nullptr, F_, W1l, b1l, W1r, F_, H_, N_, x1);

  k_agg_wmma<<<dim3(N_ / 64, 2, B_), 128, 0, stream>>>(Ab1, N_ / 32, N_,
      x1, H_, H_, agg, H_, nullptr);

  k_score<<<(B_ * N_ + 255) / 256, 256, 0, stream>>>(agg, H_, x1, H_,
      Wp1rel, bp1rel, Wp1root, H_, N_, score);

  k_topk<<<B_, 512, 0, stream>>>(score, N_, K1_, x1, H_, x1g,
      Ab1, N_ / 32, Ab2, K1_ / 32);

  k_popc<<<(B_ * K1_ + 255) / 256, 256, 0, stream>>>(Ab2, deg2, K1_, K1_ / 32);

  k_agg_wmma<<<dim3(K1_ / 64, 2, B_), 128, 0, stream>>>(Ab2, K1_ / 32, K1_,
      x1g, H_, H_, agg, H_, deg2);

  k_sage_lin<<<(B_ * K1_ * O_ + 255) / 256, 256, 0, stream>>>(agg, H_,
      nullptr, x1g, H_, W2l, b2l, W2r, H_, O_, K1_, x2);

  k_agg_wmma<<<dim3(K1_ / 64, 1, B_), 128, 0, stream>>>(Ab2, K1_ / 32, K1_,
      x2, O_, O_, agg, O_, nullptr);

  k_score<<<(B_ * K1_ + 255) / 256, 256, 0, stream>>>(agg, O_, x2, O_,
      Wp2rel, bp2rel, Wp2root, O_, K1_, score);

  k_topk<<<B_, 512, 0, stream>>>(score, K1_, K2_, x2, O_, x2g,
      Ab2, K1_ / 32, nullptr, 0);

  k_final<<<B_, 512, 0, stream>>>(x2g,
      fc1_w, fc1_b, bn1_g, bn1_b, bn1_m, bn1_v,
      fc2_w, fc2_b, bn2_g, bn2_b, bn2_m, bn2_v,
      fc3_w, fc3_b, (float*)d_out);
}
